// GroupedGlobalDeformAgg_70214125355011
// MI455X (gfx1250) — compile-verified, matrix operand order checked by reading, not run on hardware
//
#include <hip/hip_runtime.h>
#include <math.h>

typedef __attribute__((ext_vector_type(16))) __bf16 v16bf;
typedef __attribute__((ext_vector_type(8)))  float  v8f;

#define HH    64
#define WW    64
#define HWPIX 4096
#define TT    5
#define CC_   64
#define GCP_  80

// ---------------------------------------------------------------------------
// CDNA5 async global->LDS copy (ASYNCcnt-tracked) helpers.
// LDS byte address = low 32 bits of the generic pointer (flat LDS aperture).
// ---------------------------------------------------------------------------
__device__ __forceinline__ void async_copy_b128(void* lds_ptr, const void* gptr) {
  unsigned lds_addr = (unsigned)(unsigned long long)lds_ptr;
  asm volatile("global_load_async_to_lds_b128 %0, %1, off"
               :: "v"(lds_addr), "v"(gptr)
               : "memory");
}
__device__ __forceinline__ void wait_async0() {
#if __has_builtin(__builtin_amdgcn_s_wait_asynccnt)
  __builtin_amdgcn_s_wait_asynccnt(0);
#else
  asm volatile("s_wait_asynccnt 0x0" ::: "memory");
#endif
}

// K index held by fragment element e (0..15) for half-group hg (lane>>4),
// per CDNA5 ISA 16-bit A/B matrix VGPR layout (16x32 / 32x16, wave32):
//   VGPR v (=e>>1): K = (v%4)*2 + (v/4)*16 + hg*8 + (e&1)
__device__ __forceinline__ int k_of_elem(int e, int hg) {
  int v = e >> 1;
  return ((v & 3) << 1) + ((v >> 2) << 4) + (hg << 3) + (e & 1);
}

// ---------------------------------------------------------------------------
// x (B*T, C, H, W) fp32 NCHW  ->  xr bf16 NHWC group-major: [b*G+g][h][w][t*16+cc]
// ---------------------------------------------------------------------------
__global__ void rearrange_x_kernel(const float* __restrict__ x,
                                   __bf16* __restrict__ xr, int total) {
  for (int i = blockIdx.x * blockDim.x + threadIdx.x; i < total;
       i += gridDim.x * blockDim.x) {
    const int tc = i % GCP_;
    const int p  = (i / GCP_) % HWPIX;
    const int bg = i / (GCP_ * HWPIX);
    const int b = bg >> 2, g = bg & 3;
    const int t = tc >> 4, cc = tc & 15;
    xr[i] = (__bf16)x[((size_t)((b * TT + t) * CC_ + (g << 4) + cc)) * HWPIX + p];
  }
}

// ---------------------------------------------------------------------------
// Pack OIHW fp32 weights into bf16 A-fragment order:
//   pack[tap][s][ot][lane][e], A[M][K] = W[ot*16+M][s*32+K][tap]
// ---------------------------------------------------------------------------
__global__ void pack_weights_kernel(const float* __restrict__ w,
                                    __bf16* __restrict__ out,
                                    int O, int Cin, int S, int OT) {
  const int total = 9 * S * OT * 512;
  for (int i = blockIdx.x * blockDim.x + threadIdx.x; i < total;
       i += gridDim.x * blockDim.x) {
    const int e    = i & 15;
    const int lane = (i >> 4) & 31;
    const int ot   = (i >> 9) % OT;
    const int s    = (i / (512 * OT)) % S;
    const int tap  = i / (512 * OT * S);
    const int M = lane & 15, hg = lane >> 4;
    const int K  = k_of_elem(e, hg);
    const int oc = (ot << 4) + M;
    const int c  = (s << 5) + K;
    float v = 0.0f;
    if (oc < O && c < Cin) v = w[((size_t)(oc * Cin + c)) * 9 + tap];
    out[i] = (__bf16)v;
  }
}

// ---------------------------------------------------------------------------
// Generic 3x3 same-conv implicit GEMM on v_wmma_f32_16x16x32_bf16.
// Block: 32*OT threads = OT waves; one 16-pixel row strip; wave w -> out-ch tile w.
// Per tap: the tap's weight-fragment slice (S*OT KB <= 15KB) is async-copied
// global->LDS (overlapped with the B-tile im2col gather), then A and B both come
// from LDS for the WMMAs.
// Epilogues: 0 = leakyReLU(0.2) -> bf16 NHWC
//            1 = xe residual    -> fp32 NCHW + bf16 NHWC (de-rearranged)
//            2 = linear         -> fp32 NHWC   (offsets)
//            3 = 2*sigmoid      -> fp32 NHWC   (masks)
// ---------------------------------------------------------------------------
template <int EPI>
__global__ void conv3x3_wmma_kernel(const __bf16* __restrict__ in,
                                    const __bf16* __restrict__ wpack,
                                    const float* __restrict__ bias,
                                    int Cin, int S, int Co, int OT,
                                    float* __restrict__ out_f32,
                                    __bf16* __restrict__ out_bf16,
                                    const float* __restrict__ xsrc) {
  __shared__ __align__(32) __bf16 Bt[32][16];
  __shared__ __align__(32) __bf16 Wt[7680];   // up to S=3,OT=5 -> 15KB per tap
  const int tid  = threadIdx.x;
  const int lane = tid & 31;
  const int wave = tid >> 5;
  const int img  = blockIdx.x >> 8;       // 256 pixel tiles per 64x64 image
  const int tp   = blockIdx.x & 255;
  const int h    = tp >> 2;
  const int w0   = (tp & 3) << 4;

  v8f acc = {};
  for (int tap = 0; tap < 9; ++tap) {
    const int ky = tap / 3 - 1, kx = tap % 3 - 1;
    const int ih = h + ky;
    // async-stage this tap's weight fragments into LDS (ASYNCcnt path)
    {
      const int chunks = S * OT * 64;     // 16-byte chunks
      const __bf16* wtap = wpack + (size_t)tap * S * OT * 512;
      for (int i = tid; i < chunks; i += blockDim.x)
        async_copy_b128(&Wt[i * 8], wtap + (size_t)i * 8);
    }
    for (int s = 0; s < S; ++s) {
      for (int idx = tid; idx < 512; idx += blockDim.x) {
        const int l = idx >> 4, e = idx & 15;
        const int n = l & 15, hg = l >> 4;
        const int c  = (s << 5) + k_of_elem(e, hg);
        const int iw = w0 + n + kx;
        __bf16 v = (__bf16)0.0f;
        if (c < Cin && ih >= 0 && ih < HH && iw >= 0 && iw < WW)
          v = in[((size_t)(img * HWPIX + ih * WW + iw)) * Cin + c];
        Bt[l][e] = v;
      }
      if (s == 0) wait_async0();
      __syncthreads();
      {
        v16bf b = *reinterpret_cast<const v16bf*>(&Bt[lane][0]);
        v16bf a = *reinterpret_cast<const v16bf*>(
            &Wt[((size_t)((s * OT + wave) * 32 + lane)) * 16]);
        acc = __builtin_amdgcn_wmma_f32_16x16x32_bf16(false, a, false, b,
                                                      (short)0, acc, false, false);
      }
      __syncthreads();
    }
  }

  const int px    = lane & 15;
  const int w     = w0 + px;
  const int mbase = (lane >> 4) << 3;
#pragma unroll
  for (int r = 0; r < 8; ++r) {
    const int oc = (wave << 4) + mbase + r;
    if (oc >= Co) continue;
    float v = acc[r] + bias[oc];
    if constexpr (EPI == 0) {
      v = v > 0.0f ? v : 0.2f * v;
      out_bf16[((size_t)(img * HWPIX + h * WW + w)) * Co + oc] = (__bf16)v;
    } else if constexpr (EPI == 1) {
      // de-rearrange back to (b*T+t, C) and add residual xr (a permuted view of x)
      const int b = img >> 2, g = img & 3;
      const int t = oc >> 4, cc = oc & 15;
      const int bt = b * TT + t, ch = (g << 4) + cc;
      const size_t pidx = (size_t)h * WW + w;
      const size_t nchw = ((size_t)(bt * CC_ + ch)) * HWPIX + pidx;
      v += xsrc[nchw];
      out_f32[nchw] = v;
      out_bf16[((size_t)bt * HWPIX + pidx) * CC_ + ch] = (__bf16)v;
    } else if constexpr (EPI == 2) {
      out_f32[((size_t)(img * HWPIX + h * WW + w)) * Co + oc] = v;
    } else {
      out_f32[((size_t)(img * HWPIX + h * WW + w)) * Co + oc] =
          2.0f / (1.0f + __expf(-v));
    }
  }
}

// ---------------------------------------------------------------------------
// Modulated deformable conv, fused over the temporal sum:
//   out[bt] = xe[bt] + sum_{i<5} deform(xe[bt], off_i, mask_i; reg_w) + 5*reg_b
// Block: 128 threads (4 waves) = 16 pixels x 64 out channels.
// The full reg_w fragment pack (72KB) is async-copied global->LDS once per
// block; the 5*9*2 = 90 WMMAs then read A and B entirely from LDS while the
// bilinear gathers hit the L2-resident xe image.
// ---------------------------------------------------------------------------
__global__ void deform_agg_wmma_kernel(const __bf16* __restrict__ xe_bf,
                                       const float* __restrict__ xe_f32,
                                       const float* __restrict__ off,
                                       const float* __restrict__ mod,
                                       const __bf16* __restrict__ wpack,
                                       const float* __restrict__ reg_b,
                                       float* __restrict__ out) {
  __shared__ __align__(32) __bf16 Bt[32][16];
  __shared__ __align__(32) __bf16 Wlds[36864];   // 9*2*4*512 bf16 = 72KB
  __shared__ float sy[16], sx[16], sm[16];
  const int tid  = threadIdx.x;          // 128
  const int lane = tid & 31;
  const int wave = tid >> 5;
  const int bt   = blockIdx.y;
  const int tp   = blockIdx.x;           // 0..255
  const int h    = tp >> 2;
  const int w0   = (tp & 3) << 4;
  const __bf16* img = xe_bf + (size_t)bt * HWPIX * CC_;

  // stage the whole deform weight pack into LDS via the async-copy engine
  for (int i = tid; i < 4608; i += 128)            // 4608 x 16B = 72KB
    async_copy_b128(&Wlds[i * 8], wpack + (size_t)i * 8);

  if (tid == 0) {
    __builtin_prefetch(off + ((size_t)bt * HWPIX + h * WW + w0) * 90, 0, 1);
    __builtin_prefetch(mod + ((size_t)bt * HWPIX + h * WW + w0) * 45, 0, 1);
  }
  wait_async0();
  __syncthreads();

  v8f acc = {};
  for (int i = 0; i < TT; ++i) {
    for (int tap = 0; tap < 9; ++tap) {
      const int ky = tap / 3 - 1, kx = tap % 3 - 1;
      if (tid < 16) {
        const size_t p = (size_t)bt * HWPIX + h * WW + (w0 + tid);
        const float dy = off[p * 90 + i * 18 + 2 * tap];
        const float dx = off[p * 90 + i * 18 + 2 * tap + 1];
        sm[tid] = mod[p * 45 + i * 9 + tap];
        sy[tid] = (float)(h + ky) + dy;
        sx[tid] = (float)(w0 + tid + kx) + dx;
      }
      __syncthreads();
      for (int s = 0; s < 2; ++s) {
        for (int idx = tid; idx < 512; idx += 128) {
          const int l = idx >> 4, e = idx & 15;
          const int n = l & 15, hg = l >> 4;
          const int c = (s << 5) + k_of_elem(e, hg);
          const float py = sy[n], px = sx[n];
          const float y0f = floorf(py), x0f = floorf(px);
          const int y0 = (int)y0f, x0 = (int)x0f;
          const float wy1 = py - y0f, wx1 = px - x0f;
          float v = 0.0f;
#pragma unroll
          for (int cy = 0; cy < 2; ++cy) {
            const int yy = y0 + cy;
            const float wy = cy ? wy1 : 1.0f - wy1;
#pragma unroll
            for (int cx = 0; cx < 2; ++cx) {
              const int xx = x0 + cx;
              const float wx = cx ? wx1 : 1.0f - wx1;
              if (yy >= 0 && yy < HH && xx >= 0 && xx < WW)
                v += wy * wx * (float)img[((size_t)yy * WW + xx) * CC_ + c];
            }
          }
          Bt[l][e] = (__bf16)(v * sm[n]);
        }
        __syncthreads();
        {
          v16bf b = *reinterpret_cast<const v16bf*>(&Bt[lane][0]);
          v16bf a = *reinterpret_cast<const v16bf*>(
              &Wlds[((size_t)(((tap * 2 + s) * 4 + wave) * 32 + lane)) * 16]);
          acc = __builtin_amdgcn_wmma_f32_16x16x32_bf16(false, a, false, b,
                                                        (short)0, acc, false, false);
        }
        __syncthreads();
      }
    }
  }

  const int px    = lane & 15;
  const int w     = w0 + px;
  const int mbase = (lane >> 4) << 3;
#pragma unroll
  for (int r = 0; r < 8; ++r) {
    const int oc = (wave << 4) + mbase + r;
    const size_t a = ((size_t)(bt * CC_ + oc)) * HWPIX + h * WW + w;
    out[a] = xe_f32[a] + acc[r] + 5.0f * reg_b[oc];
  }
}

// ---------------------------------------------------------------------------
extern "C" void kernel_launch(void* const* d_in, const int* in_sizes, int n_in,
                              void* d_out, int out_size, void* d_ws, size_t ws_size,
                              hipStream_t stream) {
  (void)in_sizes; (void)n_in; (void)out_size; (void)ws_size;
  const float* x      = (const float*)d_in[0];
  const float* gcr_w1 = (const float*)d_in[1];
  const float* gcr_b1 = (const float*)d_in[2];
  const float* gcr_w2 = (const float*)d_in[3];
  const float* gcr_b2 = (const float*)d_in[4];
  const float* off_w  = (const float*)d_in[5];
  const float* off_b  = (const float*)d_in[6];
  const float* mod_w  = (const float*)d_in[7];
  const float* mod_b  = (const float*)d_in[8];
  const float* reg_w  = (const float*)d_in[9];
  const float* reg_b  = (const float*)d_in[10];
  float* out = (float*)d_out;

  char* ws = (char*)d_ws;
  size_t cur = 0;
  auto wsalloc = [&](size_t bytes) -> void* {
    void* p = ws + cur;
    cur = (cur + bytes + 255) & ~(size_t)255;
    return p;
  };

  __bf16* xr_bf  = (__bf16*)wsalloc((size_t)16 * HWPIX * GCP_ * sizeof(__bf16));
  __bf16* h1_bf  = (__bf16*)wsalloc((size_t)16 * HWPIX * GCP_ * sizeof(__bf16));
  __bf16* xe_bf  = (__bf16*)wsalloc((size_t)20 * HWPIX * CC_ * sizeof(__bf16));
  float*  xe_f   = (float*)wsalloc((size_t)20 * HWPIX * CC_ * sizeof(float));
  float*  off_f  = (float*)wsalloc((size_t)20 * HWPIX * 90 * sizeof(float));
  float*  mod_f  = (float*)wsalloc((size_t)20 * HWPIX * 45 * sizeof(float));
  __bf16* pk_g1  = (__bf16*)wsalloc((size_t)9 * 3 * 5 * 512 * sizeof(__bf16));
  __bf16* pk_g2  = (__bf16*)wsalloc((size_t)9 * 3 * 5 * 512 * sizeof(__bf16));
  __bf16* pk_off = (__bf16*)wsalloc((size_t)9 * 2 * 6 * 512 * sizeof(__bf16));
  __bf16* pk_mod = (__bf16*)wsalloc((size_t)9 * 2 * 3 * 512 * sizeof(__bf16));
  __bf16* pk_reg = (__bf16*)wsalloc((size_t)9 * 2 * 4 * 512 * sizeof(__bf16));

  // 1) rearrange x -> group-major NHWC bf16
  {
    const int total = 16 * HWPIX * GCP_;
    rearrange_x_kernel<<<(total + 255) / 256, 256, 0, stream>>>(x, xr_bf, total);
  }
  // 2) pack weights into WMMA A-fragment order
  pack_weights_kernel<<<(9 * 3 * 5 * 512 + 255) / 256, 256, 0, stream>>>(gcr_w1, pk_g1, 80, 80, 3, 5);
  pack_weights_kernel<<<(9 * 3 * 5 * 512 + 255) / 256, 256, 0, stream>>>(gcr_w2, pk_g2, 80, 80, 3, 5);
  pack_weights_kernel<<<(9 * 2 * 6 * 512 + 255) / 256, 256, 0, stream>>>(off_w, pk_off, 90, 64, 2, 6);
  pack_weights_kernel<<<(9 * 2 * 3 * 512 + 255) / 256, 256, 0, stream>>>(mod_w, pk_mod, 45, 64, 2, 3);
  pack_weights_kernel<<<(9 * 2 * 4 * 512 + 255) / 256, 256, 0, stream>>>(reg_w, pk_reg, 64, 64, 2, 4);

  // 3) gcr conv1 + leakyReLU (16 imgs, 80->80, 3 K-slices, 5 oc tiles)
  conv3x3_wmma_kernel<0><<<dim3(16 * 256), dim3(160), 0, stream>>>(
      xr_bf, pk_g1, gcr_b1, 80, 3, 80, 5, nullptr, h1_bf, nullptr);
  // 4) gcr conv2 + residual -> xe (fp32 NCHW + bf16 NHWC, de-rearranged)
  conv3x3_wmma_kernel<1><<<dim3(16 * 256), dim3(160), 0, stream>>>(
      h1_bf, pk_g2, gcr_b2, 80, 3, 80, 5, xe_f, xe_bf, x);
  // 5) offset conv (20 imgs, 64->90)
  conv3x3_wmma_kernel<2><<<dim3(20 * 256), dim3(192), 0, stream>>>(
      xe_bf, pk_off, off_b, 64, 2, 90, 6, off_f, nullptr, nullptr);
  // 6) mask conv + 2*sigmoid (20 imgs, 64->45)
  conv3x3_wmma_kernel<3><<<dim3(20 * 256), dim3(96), 0, stream>>>(
      xe_bf, pk_mod, mod_b, 64, 2, 45, 3, mod_f, nullptr, nullptr);
  // 7) fused deformable conv + temporal sum + residual -> out
  deform_agg_wmma_kernel<<<dim3(256, 20), dim3(128), 0, stream>>>(
      xe_bf, xe_f, off_f, mod_f, pk_reg, reg_b, out);
}
